// GAT_50732153700397
// MI455X (gfx1250) — matrix operand order repaired and checked by reading, not compiled, not run on hardware
//
#include <hip/hip_runtime.h>
#include <hip/hip_bf16.h>
#include <math.h>

// Problem constants (from reference setup_inputs)
#define BATCH 8
#define NN    2048
#define DD    256
#define KP    (DD / 2)     // 128 bf16-pairs along K
#define AST   132          // padded u32 stride: 132%64=4 -> conflict-free, 16B-aligned rows

typedef __attribute__((ext_vector_type(16))) __bf16        v16bf;
typedef __attribute__((ext_vector_type(8)))  float         v8f;
typedef __attribute__((ext_vector_type(4)))  unsigned int  u32x4;

union FragBF {
    unsigned int u[8];
    u32x4        q[2];
    v16bf        v;
};

// ---- LDS layout (bytes) ----
// S    : float[16][2048]                 131072
// Ahi/Alo : u32[16*132] each               2*8448
// Bhi/Blo : u32[8][16*132] each           2*67584
// Emax : u32[16], Tmax: u32[16], Ssum: float[16]  192
#define SMEM_BYTES (131072 + 2*8448 + 2*67584 + 192)

__device__ __forceinline__ unsigned int mono(float f) {
    unsigned int u = __float_as_uint(f);
    return (u & 0x80000000u) ? ~u : (u | 0x80000000u);
}
__device__ __forceinline__ float demono(unsigned int u) {
    u = (u & 0x80000000u) ? (u & 0x7FFFFFFFu) : ~u;
    return __uint_as_float(u);
}

// Split fp32 into bf16 hi (RNE) + bf16 lo (residual), returning 16-bit patterns.
__device__ __forceinline__ void split_bf16(float x, unsigned int& hi, unsigned int& lo) {
    unsigned int xb = __float_as_uint(x);
    unsigned int h16 = (xb + 0x7FFFu + ((xb >> 16) & 1u)) >> 16;   // bf16 RNE
    float hf = __uint_as_float(h16 << 16);
    float l  = x - hf;
    unsigned int lb = __float_as_uint(l);
    unsigned int l16 = (lb + 0x7FFFu + ((lb >> 16) & 1u)) >> 16;
    hi = h16 & 0xFFFFu;
    lo = l16 & 0xFFFFu;
}

// One-time fp32 -> packed bf16 hi/lo conversion of h into workspace.
// Each thread handles one u64-chunk (4 floats -> 2 packed u32 per component).
__global__ __launch_bounds__(256) void gat_convert_kernel(const float* __restrict__ h,
                                                          unsigned int* __restrict__ whi,
                                                          unsigned int* __restrict__ wlo) {
    const size_t c = (size_t)blockIdx.x * 256 + threadIdx.x;   // 0 .. B*N*D/4-1
    const float4 x = ((const float4*)h)[c];
    unsigned int h0, l0, h1, l1, h2, l2, h3, l3;
    split_bf16(x.x, h0, l0);
    split_bf16(x.y, h1, l1);
    split_bf16(x.z, h2, l2);
    split_bf16(x.w, h3, l3);
    uint2 hv, lv;
    hv.x = h0 | (h1 << 16);  hv.y = h2 | (h3 << 16);
    lv.x = l0 | (l1 << 16);  lv.y = l2 | (l3 << 16);
    ((uint2*)whi)[c] = hv;
    ((uint2*)wlo)[c] = lv;
}

__global__ __launch_bounds__(256, 1) void gat_fused_kernel(const float* __restrict__ h,
                                                           const float* __restrict__ adj,
                                                           float* __restrict__ out,
                                                           const unsigned int* __restrict__ hbhi,
                                                           const unsigned int* __restrict__ hblo) {
    extern __shared__ __align__(16) char smem[];
    float*        S    = (float*)smem;                                  // [16][2048]
    unsigned int* Ahi  = (unsigned int*)(smem + 131072);
    unsigned int* Alo  = Ahi + 16 * AST;
    unsigned int* BhiA = Alo + 16 * AST;                                // [8][16*AST]
    unsigned int* BloA = BhiA + 8 * 16 * AST;
    unsigned int* Emax = BloA + 8 * 16 * AST;                           // [16]
    unsigned int* Tmax = Emax + 16;                                     // [16]
    float*        Ssum = (float*)(Tmax + 16);                           // [16]

    const int tid  = threadIdx.x;
    const int wave = tid >> 5;
    const int lane = tid & 31;
    const int b    = blockIdx.x >> 7;
    const int i0   = (blockIdx.x & 127) * 16;

    const float* hb  = h + (size_t)b * NN * DD;
    const bool   pre = (hbhi != nullptr);

    if (tid < 16) { Emax[tid] = 0u; Tmax[tid] = 0u; Ssum[tid] = 0.0f; }

    // ---- Stage A panel: rows i0..i0+15 as packed bf16 hi/lo pairs ----
    if (pre) {
        const size_t rowbase = ((size_t)b * NN + i0) * KP;
        for (int o = tid; o < 16 * 32; o += 256) {
            const int r = o >> 5, q = (o & 31) * 4;
            *(u32x4*)&Ahi[r * AST + q] = *(const u32x4*)&hbhi[rowbase + (size_t)r * KP + q];
            *(u32x4*)&Alo[r * AST + q] = *(const u32x4*)&hblo[rowbase + (size_t)r * KP + q];
        }
    } else {
        for (int o = tid; o < 16 * KP; o += 256) {
            const int r = o >> 7, kp = o & (KP - 1);
            float2 xy = ((const float2*)(hb + (size_t)(i0 + r) * DD))[kp];
            unsigned int hx, lx, hy, ly;
            split_bf16(xy.x, hx, lx);
            split_bf16(xy.y, hy, ly);
            Ahi[r * AST + kp] = hx | (hy << 16);
            Alo[r * AST + kp] = lx | (ly << 16);
        }
    }
    __syncthreads();

    // ---- Hoist A fragments for all 8 K-steps into registers (128 VGPRs) ----
    const int mn = lane & 15;      // A row / B col within tile
    const int hl = lane >> 4;      // lane half
    FragBF a_hi[8], a_lo[8];
#pragma unroll
    for (int ks = 0; ks < 8; ++ks) {
        // A 16x32 bf16 layout: vgpr v holds K-pair (v>>2)*8 + half*4 + (v&3)
        const int base = mn * AST + ks * 16 + hl * 4;
        a_hi[ks].q[0] = *(const u32x4*)&Ahi[base];
        a_hi[ks].q[1] = *(const u32x4*)&Ahi[base + 8];
        a_lo[ks].q[0] = *(const u32x4*)&Alo[base];
        a_lo[ks].q[1] = *(const u32x4*)&Alo[base + 8];
    }

    // ---- Pass 1: energy strip via bf16x3 WMMA ----
    unsigned int* bh = BhiA + wave * (16 * AST);
    unsigned int* bl = BloA + wave * (16 * AST);

    for (int jt = wave; jt < NN / 16; jt += 8) {
        const int j0 = jt * 16;
        // Stage this wave's B tile (rows j0..j0+15)
        if (pre) {
            const size_t rowbase = ((size_t)b * NN + j0) * KP;
            for (int o = lane; o < 16 * 32; o += 32) {
                const int r = o >> 5, q = (o & 31) * 4;
                *(u32x4*)&bh[r * AST + q] = *(const u32x4*)&hbhi[rowbase + (size_t)r * KP + q];
                *(u32x4*)&bl[r * AST + q] = *(const u32x4*)&hblo[rowbase + (size_t)r * KP + q];
            }
        } else {
            for (int o = lane; o < 16 * KP; o += 32) {
                const int r = o >> 7, kp = o & (KP - 1);
                float2 xy = ((const float2*)(hb + (size_t)(j0 + r) * DD))[kp];
                unsigned int hx, lx, hy, ly;
                split_bf16(xy.x, hx, lx);
                split_bf16(xy.y, hy, ly);
                bh[r * AST + kp] = hx | (hy << 16);
                bl[r * AST + kp] = lx | (ly << 16);
            }
        }
        asm volatile("s_wait_dscnt 0" ::: "memory");   // wave-local stage->gather ordering

        v8f c = {0.f, 0.f, 0.f, 0.f, 0.f, 0.f, 0.f, 0.f};
#pragma unroll
        for (int ks = 0; ks < 8; ++ks) {
            FragBF b_hi, b_lo;
            // B 32x16 bf16 layout: vgpr v holds K-pair (v>>2)*8 + half*4 + (v&3)
            const int bbase = mn * AST + ks * 16 + hl * 4;
            b_hi.q[0] = *(const u32x4*)&bh[bbase];
            b_hi.q[1] = *(const u32x4*)&bh[bbase + 8];
            b_lo.q[0] = *(const u32x4*)&bl[bbase];
            b_lo.q[1] = *(const u32x4*)&bl[bbase + 8];

            c = __builtin_amdgcn_wmma_f32_16x16x32_bf16(false, a_hi[ks].v, false, b_hi.v,
                                                        (short)0, c, false, false);
            c = __builtin_amdgcn_wmma_f32_16x16x32_bf16(false, a_hi[ks].v, false, b_lo.v,
                                                        (short)0, c, false, false);
            c = __builtin_amdgcn_wmma_f32_16x16x32_bf16(false, a_lo[ks].v, false, b_hi.v,
                                                        (short)0, c, false, false);
        }
        // C/D layout: vgpr g -> row g + 8*half, col = lane&15
#pragma unroll
        for (int g = 0; g < 8; ++g) {
            S[(g + 8 * hl) * NN + j0 + mn] = c[g];
        }
    }
    __syncthreads();

    const int r  = tid >> 4;         // row 0..15
    const int ch = tid & 15;         // 128-col chunk 0..15
    float* srow = S + r * NN + ch * 128;

    // ---- Pass 1.5: row max of energy ----
    {
        float m = -INFINITY;
        for (int i = 0; i < 128; ++i) m = fmaxf(m, srow[i]);
        atomicMax(&Emax[r], mono(m));
    }
    __syncthreads();

    // ---- Pass 2: t = (max_e - e) * adj, track row max of t ----
    {
        const float me = demono(Emax[r]);
        const float* arow = adj + (size_t)b * NN * NN + (size_t)(i0 + r) * NN + ch * 128;
        float tm = 0.0f;   // t >= 0 always (adj in [0,1], max_e >= e)
        for (int i = 0; i < 128; i += 4) {
            float4 a4 = *(const float4*)(arow + i);
            float4 e4 = *(const float4*)(srow + i);
            float4 t4;
            t4.x = (me - e4.x) * a4.x;
            t4.y = (me - e4.y) * a4.y;
            t4.z = (me - e4.z) * a4.z;
            t4.w = (me - e4.w) * a4.w;
            tm = fmaxf(tm, fmaxf(fmaxf(t4.x, t4.y), fmaxf(t4.z, t4.w)));
            *(float4*)(srow + i) = t4;
        }
        atomicMax(&Tmax[r], mono(tm));
    }
    __syncthreads();

    // ---- Pass 3: exp(t - max_t), row sums ----
    {
        const float tmx = demono(Tmax[r]);
        float s = 0.0f;
        for (int i = 0; i < 128; i += 4) {
            float4 t4 = *(const float4*)(srow + i);
            t4.x = __expf(t4.x - tmx);
            t4.y = __expf(t4.y - tmx);
            t4.z = __expf(t4.z - tmx);
            t4.w = __expf(t4.w - tmx);
            s += (t4.x + t4.y) + (t4.z + t4.w);
            *(float4*)(srow + i) = t4;
        }
        atomicAdd(&Ssum[r], s);
    }
    __syncthreads();

    // ---- Pass 4: normalize, coalesced float4 stores ----
    {
        float* ob = out + (size_t)b * NN * NN + (size_t)i0 * NN;
        for (int rr = 0; rr < 16; ++rr) {
            const float inv = 1.0f / Ssum[rr];
            float4*       dst = (float4*)(ob + (size_t)rr * NN);
            const float4* src = (const float4*)(S + rr * NN);
#pragma unroll
            for (int s2 = 0; s2 < 2; ++s2) {
                const int idx = tid + s2 * 256;     // float4 index 0..511
                float4 v = src[idx];
                v.x *= inv; v.y *= inv; v.z *= inv; v.w *= inv;
                dst[idx] = v;
            }
        }
    }
}

extern "C" void kernel_launch(void* const* d_in, const int* in_sizes, int n_in,
                              void* d_out, int out_size, void* d_ws, size_t ws_size,
                              hipStream_t stream) {
    const float* h   = (const float*)d_in[0];
    const float* adj = (const float*)d_in[1];
    float*       out = (float*)d_out;

    (void)in_sizes; (void)n_in; (void)out_size;

    const size_t comp_elems = (size_t)BATCH * NN * KP;      // packed u32 per component
    const size_t ws_need    = comp_elems * 4u * 2u;         // 16 MB

    const unsigned int* hbhi = nullptr;
    const unsigned int* hblo = nullptr;
    if (d_ws != nullptr && ws_size >= ws_need) {
        unsigned int* whi = (unsigned int*)d_ws;
        unsigned int* wlo = whi + comp_elems;
        const int total_chunks = BATCH * NN * DD / 4;       // 1,048,576
        gat_convert_kernel<<<dim3(total_chunks / 256), dim3(256), 0, stream>>>(h, whi, wlo);
        hbhi = whi;
        hblo = wlo;
    }

    hipFuncSetAttribute((const void*)gat_fused_kernel,
                        hipFuncAttributeMaxDynamicSharedMemorySize, SMEM_BYTES);

    dim3 grid(BATCH * (NN / 16));   // 1024 workgroups: (batch, 16-row tile)
    dim3 block(256);                // 8 waves of 32
    gat_fused_kernel<<<grid, block, SMEM_BYTES, stream>>>(h, adj, out, hbhi, hblo);
}
